// TinyMiniMindCLIPContrastiveVLM_52278341927624
// MI455X (gfx1250) — matrix operand order repaired and checked by reading, not compiled, not run on hardware
//
#include <hip/hip_runtime.h>
#include <math.h>

typedef __attribute__((ext_vector_type(16))) __bf16        bf16x16;
typedef __attribute__((ext_vector_type(8)))  float         f32x8;
typedef __attribute__((ext_vector_type(4)))  unsigned int  u32x4;
typedef __attribute__((ext_vector_type(2)))  unsigned int  u32x2;

#define B_    64
#define P_    256
#define DV    1024
#define DT    768
#define S_    512
#define NSEG  32
#define MSEG  (B_ * NSEG)   /* 2048 rows after segment pre-reduction */

__device__ __forceinline__ unsigned short f2bf(float f) {
  unsigned u = __float_as_uint(f);
  unsigned r = u + 0x7FFFu + ((u >> 16) & 1u);   // round-to-nearest-even
  return (unsigned short)(r >> 16);
}
__device__ __forceinline__ float bf2f(unsigned short h) {
  return __uint_as_float(((unsigned)h) << 16);
}
// split (a,b) into packed bf16 hi-pair and lo-pair words
struct HL { unsigned hi, lo; };
__device__ __forceinline__ HL split2(float a, float b) {
  const unsigned short h0 = f2bf(a), h1 = f2bf(b);
  const unsigned short l0 = f2bf(a - bf2f(h0)), l1 = f2bf(b - bf2f(h1));
  HL r;
  r.hi = (unsigned)h0 | ((unsigned)h1 << 16);
  r.lo = (unsigned)l0 | ((unsigned)l1 << 16);
  return r;
}

// ---------------------------------------------------------------------------
// Kernel 1: fused LayerNorm (stop-grad) + per-segment sum over 8 patches.
// One block per (image, segment). Single pass over vis_feats (67 MB).
// Rseg[b*32+s][k] = sum_{j=0..7} ( (x-mu)*rstd*gamma[k] ) + 8*beta[k]
// ---------------------------------------------------------------------------
__global__ void __launch_bounds__(256)
seg_ln_reduce(const float* __restrict__ vis, const float* __restrict__ gamma,
              const float* __restrict__ beta, float* __restrict__ Rseg) {
  const int blk  = blockIdx.x;        // b*32 + s ; global patch rows blk*8..blk*8+7
  const int t    = threadIdx.x;       // 256 threads, 4 cols each
  const int lane = t & 31, wv = t >> 5;
  __shared__ float red1[8][8], red2[8][8];
  __shared__ float smu[8], srs[8];

  float x[8][4];
  const size_t rowbase = (size_t)blk * 8;
  for (int j = 0; j < 8; ++j) {
    const float4 v = *(const float4*)(vis + (rowbase + j) * DV + t * 4);
    x[j][0] = v.x; x[j][1] = v.y; x[j][2] = v.z; x[j][3] = v.w;
    float s1 = v.x + v.y + v.z + v.w;
    float s2 = v.x * v.x + v.y * v.y + v.z * v.z + v.w * v.w;
    for (int o = 16; o > 0; o >>= 1) {          // wave32 reduction
      s1 += __shfl_xor(s1, o);
      s2 += __shfl_xor(s2, o);
    }
    if (lane == 0) { red1[j][wv] = s1; red2[j][wv] = s2; }
  }
  __syncthreads();
  if (t < 8) {
    float S1 = 0.f, S2 = 0.f;
    for (int w = 0; w < 8; ++w) { S1 += red1[t][w]; S2 += red2[t][w]; }
    const float mu  = S1 * (1.0f / DV);
    const float var = S2 * (1.0f / DV) - mu * mu;   // biased var (ddof=0)
    smu[t] = mu;
    srs[t] = rsqrtf(var + 1e-5f);
  }
  __syncthreads();
  float mu[8], rs[8];
  for (int j = 0; j < 8; ++j) { mu[j] = smu[j]; rs[j] = srs[j]; }
  for (int c = 0; c < 4; ++c) {
    float a = 0.f;
    for (int j = 0; j < 8; ++j) a += (x[j][c] - mu[j]) * rs[j];
    const int k = t * 4 + c;
    Rseg[(size_t)blk * DV + k] = a * gamma[k] + 8.0f * beta[k];
  }
}

// ---------------------------------------------------------------------------
// Kernel 2: seg_sum[2048,768] = Rseg[2048,1024] @ W[1024,768] + 8*b_proj
// bf16 hi/lo split (3 WMMAs ~ fp32 accuracy), v_wmma_f32_16x16x32_bf16.
// Block tile 64(M) x 128(N), BK=32, 256 threads = 8 waves, each wave 16x64.
// B tile is micro-transposed 4x4 in registers so all LDS stores are packed
// 8-byte runs along K (no scalar b16 LDS traffic).
// ---------------------------------------------------------------------------
union Frag { u32x4 q[2]; bf16x16 v; };

__global__ void __launch_bounds__(256)
wmma_gemm(const float* __restrict__ A, const float* __restrict__ W,
          const float* __restrict__ bias, float* __restrict__ C) {
  __shared__ __attribute__((aligned(16))) unsigned short sAh[64 * 32];
  __shared__ __attribute__((aligned(16))) unsigned short sAl[64 * 32];
  __shared__ __attribute__((aligned(16))) unsigned short sBh[128 * 32];
  __shared__ __attribute__((aligned(16))) unsigned short sBl[128 * 32];

  const int t    = threadIdx.x;
  const int m0   = blockIdx.y * 64;
  const int n0   = blockIdx.x * 128;
  const int wv   = t >> 5, lane = t & 31;
  const int wm   = (wv & 3) * 16;          // wave M block
  const int wn   = (wv >> 2) * 64;         // wave N half
  const int lidx = lane & 15;

  const f32x8 vzero = {0.f, 0.f, 0.f, 0.f, 0.f, 0.f, 0.f, 0.f};
  f32x8 acc[4] = {vzero, vzero, vzero, vzero};

  const int arow = t >> 3;                 // 0..31 (A tile row, +32 on 2nd iter)
  const int akc  = (t & 7) * 4;            // 0..28 (A k-chunk)
  const int bnc  = (t & 31) * 4;           // 0..124 (B n-chunk)
  const int bkq  = (t >> 5) * 4;           // 0..28 (B k-quad base)

  for (int k0 = 0; k0 < DV; k0 += 32) {
    if (k0 + 32 < DV) {                    // prefetch next K-slab
      __builtin_prefetch(A + (size_t)(m0 + arow) * DV + k0 + 32 + akc, 0, 3);
      __builtin_prefetch(W + (size_t)(k0 + 32 + bkq) * DT + n0 + bnc, 0, 3);
    }
    // ---- stage A tile (64x32) as bf16 hi/lo, row-major [m][k] ----
    for (int i = 0; i < 2; ++i) {
      const int r = arow + i * 32;
      const float4 v = *(const float4*)(A + (size_t)(m0 + r) * DV + k0 + akc);
      const HL p0 = split2(v.x, v.y);
      const HL p1 = split2(v.z, v.w);
      u32x2 ph, pl;
      ph.x = p0.hi; ph.y = p1.hi;
      pl.x = p0.lo; pl.y = p1.lo;
      *(u32x2*)&sAh[r * 32 + akc] = ph;
      *(u32x2*)&sAl[r * 32 + akc] = pl;
    }
    // ---- stage B tile (32x128) transposed to [n][k] via 4x4 register
    //      micro-transpose; packed 8-byte LDS stores along K ----
    {
      const float4 v0 = *(const float4*)(W + (size_t)(k0 + bkq + 0) * DT + n0 + bnc);
      const float4 v1 = *(const float4*)(W + (size_t)(k0 + bkq + 1) * DT + n0 + bnc);
      const float4 v2 = *(const float4*)(W + (size_t)(k0 + bkq + 2) * DT + n0 + bnc);
      const float4 v3 = *(const float4*)(W + (size_t)(k0 + bkq + 3) * DT + n0 + bnc);
      const float col[4][4] = {{v0.x, v1.x, v2.x, v3.x},
                               {v0.y, v1.y, v2.y, v3.y},
                               {v0.z, v1.z, v2.z, v3.z},
                               {v0.w, v1.w, v2.w, v3.w}};
      for (int c = 0; c < 4; ++c) {
        const HL p0 = split2(col[c][0], col[c][1]);
        const HL p1 = split2(col[c][2], col[c][3]);
        u32x2 ph, pl;
        ph.x = p0.hi; ph.y = p1.hi;
        pl.x = p0.lo; pl.y = p1.lo;
        *(u32x2*)&sBh[(bnc + c) * 32 + bkq] = ph;
        *(u32x2*)&sBl[(bnc + c) * 32 + bkq] = pl;
      }
    }
    __syncthreads();

    // A fragment: lane<16 -> M=lidx, K {0..7,16..23}; lane>=16 -> K {8..15,24..31}
    Frag ah, al;
    {
      const int r  = wm + lidx;
      const int kb = (lane & 16) ? 8 : 0;
      ah.q[0] = *(const u32x4*)&sAh[r * 32 + kb];
      ah.q[1] = *(const u32x4*)&sAh[r * 32 + kb + 16];
      al.q[0] = *(const u32x4*)&sAl[r * 32 + kb];
      al.q[1] = *(const u32x4*)&sAl[r * 32 + kb + 16];
    }
    // B fragment: lane<16 -> N=lidx, K {0..7,16..23} ; lane>=16 -> K {8..15,24..31}
    const int bkb = (lane & 16) ? 8 : 0;
    for (int ni = 0; ni < 4; ++ni) {
      Frag bh, bl;
      const int cidx = wn + ni * 16 + lidx;
      bh.q[0] = *(const u32x4*)&sBh[cidx * 32 + bkb];
      bh.q[1] = *(const u32x4*)&sBh[cidx * 32 + bkb + 16];
      bl.q[0] = *(const u32x4*)&sBl[cidx * 32 + bkb];
      bl.q[1] = *(const u32x4*)&sBl[cidx * 32 + bkb + 16];
      acc[ni] = __builtin_amdgcn_wmma_f32_16x16x32_bf16(false, ah.v, false, bh.v,
                                                        (short)0, acc[ni], false, false);
      acc[ni] = __builtin_amdgcn_wmma_f32_16x16x32_bf16(false, ah.v, false, bl.v,
                                                        (short)0, acc[ni], false, false);
      acc[ni] = __builtin_amdgcn_wmma_f32_16x16x32_bf16(false, al.v, false, bh.v,
                                                        (short)0, acc[ni], false, false);
    }
    __syncthreads();
  }

  // C layout: VGPR j, lanes 0-15 -> M=j ; lanes 16-31 -> M=j+8 ; N = lane&15
  const int mbase = m0 + wm + ((lane >> 4) << 3);
  const int nbase = n0 + wn + lidx;
  for (int ni = 0; ni < 4; ++ni) {
    const int n  = nbase + ni * 16;
    const float bv = 8.0f * bias[n];   // 8 patches each received +b_proj
    for (int j = 0; j < 8; ++j)
      C[(size_t)(mbase + j) * DT + n] = acc[ni][j] + bv;
  }
}

// ---------------------------------------------------------------------------
// Kernel 3: contrastive loss. pooled = sum_s seg_sum (scale cancels in L2
// normalization), sim = nrm@nrm.T / 0.07, symmetric diagonal CE.
// ---------------------------------------------------------------------------
__global__ void __launch_bounds__(256)
loss_kernel(const float* __restrict__ segsum, float* __restrict__ pooled,
            float* __restrict__ out_loss) {
  __shared__ float sim[B_ * B_];
  __shared__ float nrm[B_];
  __shared__ float red[B_];
  const int t = threadIdx.x;

  for (int idx = t; idx < B_ * DT; idx += 256) {
    const int i = idx / DT, n = idx - i * DT;
    const float* p = segsum + (size_t)i * NSEG * DT + n;
    float s = 0.f;
    for (int sg = 0; sg < NSEG; ++sg) s += p[(size_t)sg * DT];
    pooled[idx] = s;
  }
  __syncthreads();
  if (t < B_) {
    const float* p = pooled + (size_t)t * DT;
    float s = 0.f;
    for (int n = 0; n < DT; ++n) s += p[n] * p[n];
    nrm[t] = sqrtf(s);
  }
  __syncthreads();
  for (int e = t; e < B_ * B_; e += 256) {
    const int i = e >> 6, j = e & 63;
    const float* pi = pooled + (size_t)i * DT;
    const float* pj = pooled + (size_t)j * DT;
    float d = 0.f;
    for (int n = 0; n < DT; ++n) d += pi[n] * pj[n];
    sim[e] = d / (nrm[i] * nrm[j]) * (1.0f / 0.07f);
  }
  __syncthreads();
  if (t < B_) {
    float mx = -3.4e38f, mxc = -3.4e38f;
    for (int j = 0; j < B_; ++j) {
      mx  = fmaxf(mx,  sim[t * B_ + j]);
      mxc = fmaxf(mxc, sim[j * B_ + t]);
    }
    float se = 0.f, sec = 0.f;
    for (int j = 0; j < B_; ++j) {
      se  += expf(sim[t * B_ + j] - mx);
      sec += expf(sim[j * B_ + t] - mxc);
    }
    const float diag = sim[t * B_ + t];
    red[t] = (mx + logf(se) - diag) + (mxc + logf(sec) - diag);
  }
  __syncthreads();
  if (t == 0) {
    float s = 0.f;
    for (int i = 0; i < B_; ++i) s += red[i];
    out_loss[0] = s * (1.0f / (2.0f * B_));
  }
}

// ---------------------------------------------------------------------------
// Kernel 4: text_emb. One block per token (B*S = 32768): rank = cumsum of
// placeholder mask; placeholder -> seg_sum/8 ; else -> wte[id].
// ---------------------------------------------------------------------------
__global__ void __launch_bounds__(256)
text_kernel(const int* __restrict__ ids, const float* __restrict__ wte,
            const float* __restrict__ segsum, const int* __restrict__ ph_ptr,
            float* __restrict__ out) {
  const int token = blockIdx.x;
  const int b = token >> 9;          // S_ = 512
  const int s = token & (S_ - 1);
  __shared__ int sh_valid, sh_row;
  if (threadIdx.x == 0) {
    const int ph = *ph_ptr;
    const int* row = ids + (size_t)b * S_;
    const int id = row[s];
    int cnt = 0;
    for (int j = 0; j <= s; ++j) cnt += (row[j] == ph) ? 1 : 0;
    const int rank = cnt - 1;
    const bool valid = (id == ph) && (rank < NSEG);
    int r = rank < 0 ? 0 : (rank > NSEG - 1 ? NSEG - 1 : rank);
    sh_valid = valid ? 1 : 0;
    sh_row   = valid ? (b * NSEG + r) : id;
  }
  __syncthreads();
  const int valid = sh_valid, rowi = sh_row;
  const float* src   = valid ? (segsum + (size_t)rowi * DT) : (wte + (size_t)rowi * DT);
  const float  scale = valid ? 0.125f : 1.0f;   // seg_mean = seg_sum / 8
  float* o = out + (size_t)token * DT;
  for (int c = 0; c < 3; ++c) {                 // 768 = 3 * 256
    const int n = threadIdx.x + c * 256;
    o[n] = src[n] * scale;
  }
}

// ---------------------------------------------------------------------------
extern "C" void kernel_launch(void* const* d_in, const int* in_sizes, int n_in,
                              void* d_out, int out_size, void* d_ws, size_t ws_size,
                              hipStream_t stream) {
  const float* vis   = (const float*)d_in[0];
  const float* gamma = (const float*)d_in[1];
  const float* beta  = (const float*)d_in[2];
  const float* Wp    = (const float*)d_in[3];
  const float* bp    = (const float*)d_in[4];
  const float* wte   = (const float*)d_in[5];
  const int*   ids   = (const int*)d_in[6];
  const int*   phid  = (const int*)d_in[7];

  float* out    = (float*)d_out;
  float* Rseg   = (float*)d_ws;                        // 2048*1024 f32 (8 MB)
  float* segsum = Rseg + (size_t)MSEG * DV;            // 2048*768  f32 (6 MB)
  float* pooled = segsum + (size_t)MSEG * DT;          // 64*768    f32
  float* loss   = out + (size_t)B_ * S_ * DT;          // last output element

  seg_ln_reduce<<<MSEG, 256, 0, stream>>>(vis, gamma, beta, Rseg);
  wmma_gemm<<<dim3(DT / 128, MSEG / 64), 256, 0, stream>>>(Rseg, Wp, bp, segsum);
  loss_kernel<<<1, 256, 0, stream>>>(segsum, pooled, loss);
  text_kernel<<<B_ * S_, 256, 0, stream>>>(ids, wte, segsum, phid, out);
}
